// GraphTransformerEncoder_83872121356772
// MI455X (gfx1250) — matrix operand order repaired and checked by reading, not compiled, not run on hardware
//
#include <hip/hip_runtime.h>
#include <math.h>

// ---------------------------------------------------------------------------
// Graph Transformer encoder for MI455X (gfx1250, wave32, WMMA).
// Dense projections: pre-converted bf16 activations/weights, padded rows,
// fully-unrolled K, 16x64 strip per wave -> pure load+wmma inner body.
// Softmax/LN/pooling stay fp32. Edge phase is L2-resident gather/scatter.
// ---------------------------------------------------------------------------

#define HIDDEN   128
#define HEADS    8
#define HEAD_DIM 16
#define NLAYERS  4
#define NGRAPHS  64

typedef __attribute__((ext_vector_type(16))) __bf16          v16bf;
typedef __attribute__((ext_vector_type(16))) unsigned short  v16us;
typedef __attribute__((ext_vector_type(8)))  float           v8f;

// ---- helpers --------------------------------------------------------------

__device__ __forceinline__ unsigned short f2b(float f) {
    // fp32 -> bf16, round-to-nearest-even
    unsigned u = __float_as_uint(f);
    unsigned r = u + 0x7FFFu + ((u >> 16) & 1u);
    return (unsigned short)(r >> 16);
}

// order-preserving float<->uint encoding so atomicMax(uint) == float max
__device__ __forceinline__ unsigned encF(float f) {
    unsigned u = __float_as_uint(f);
    return (u & 0x80000000u) ? ~u : (u | 0x80000000u);
}
__device__ __forceinline__ float decF(unsigned u) {
    return __uint_as_float((u & 0x80000000u) ? (u & 0x7FFFFFFFu) : ~u);
}
#define NEG_INF_ENC 0x007FFFFFu   // encF(-inf)

__device__ __forceinline__ float blockSum128(float v, float* red) {
    #pragma unroll
    for (int off = 16; off > 0; off >>= 1) v += __shfl_down(v, off, 32);
    __syncthreads();                         // protect red re-use
    if ((threadIdx.x & 31) == 0) red[threadIdx.x >> 5] = v;
    __syncthreads();
    return red[0] + red[1] + red[2] + red[3];
}

// ---- conversions (fp32 -> bf16 bits, with zero row padding) ---------------

__global__ void conv_bf16_pad(const float* __restrict__ src, unsigned short* __restrict__ dst,
                              long long n_valid, long long n_total) {
    long long i = (long long)blockIdx.x * blockDim.x + threadIdx.x;
    if (i >= n_total) return;
    dst[i] = (i < n_valid) ? f2b(src[i]) : (unsigned short)0;
}

// WallB[l][row(0..511)][k(0..127)], rows 0-127=Wq, 128-255=Wk, 256-383=Wv, 384-511=Ws
__global__ void prep_wall(const float* __restrict__ Wq, const float* __restrict__ Wk,
                          const float* __restrict__ Wv, const float* __restrict__ Ws,
                          unsigned short* __restrict__ WallB) {
    int i = blockIdx.x * blockDim.x + threadIdx.x;
    const int per = 512 * HIDDEN;
    if (i >= NLAYERS * per) return;
    int l = i / per, rem = i % per;
    int row = rem / HIDDEN, k = rem % HIDDEN;
    int mat = row >> 7, j = row & 127;
    const float* W = (mat == 0) ? Wq : (mat == 1) ? Wk : (mat == 2) ? Wv : Ws;
    WallB[i] = f2b(W[((size_t)l * HIDDEN + j) * HIDDEN + k]);
}

__global__ void prep_bias(const float* __restrict__ bq, const float* __restrict__ bk,
                          const float* __restrict__ bv, const float* __restrict__ bs,
                          float* __restrict__ ball) {
    int i = blockIdx.x * blockDim.x + threadIdx.x;
    if (i >= NLAYERS * 512) return;
    int l = i / 512, row = i % 512;
    int mat = row >> 7, j = row & 127;
    const float* b = (mat == 0) ? bq : (mat == 1) ? bk : (mat == 2) ? bv : bs;
    ball[i] = b[l * HIDDEN + j];
}

// ---- WMMA GEMM ------------------------------------------------------------
// C[Nrows,Ncols] = A[Mpad,K](bf16) @ B[Ncols,K](bf16)^T + bias
// One wave per 16x64 strip: A fragment reused across 4 N-subtiles,
// K fully unrolled -> 4*(K/32) straight-line v_wmma_f32_16x16x32_bf16.
template <int K>
__global__ __launch_bounds__(32)
void gemm_bf16_wmma(const unsigned short* __restrict__ A, const unsigned short* __restrict__ B,
                    const float* __restrict__ bias, float* __restrict__ C,
                    int Nrows, int Ncols) {
    const int lane = threadIdx.x & 31;
    const int ln = lane & 15;          // A-row / B-column selector
    const int hs = lane >> 4;          // lane-half selector
    const int m0 = blockIdx.x * 16;
    const int n0 = blockIdx.y * 64;

    const unsigned short* Arow = A + (size_t)(m0 + ln) * K;
    v8f acc[4] = {};

    #pragma unroll
    for (int kb = 0; kb < K; kb += 32) {
        // A fragment (16-bit A 16x32 layout): two contiguous runs of 8 bf16
        union { uint4 q[2]; v16us v; } ua;
        ua.q[0] = *(const uint4*)(Arow + kb + hs * 8);
        ua.q[1] = *(const uint4*)(Arow + kb + 16 + hs * 8);
        v16bf a = __builtin_bit_cast(v16bf, ua.v);

        #pragma unroll
        for (int sub = 0; sub < 4; ++sub) {
            // B fragment (32x16): two runs of 8 bf16, same k order as A
            const unsigned short* Bp = B + (size_t)(n0 + sub * 16 + ln) * K + kb + hs * 8;
            union { uint4 q[2]; v16us v; } ub;
            ub.q[0] = *(const uint4*)(Bp);
            ub.q[1] = *(const uint4*)(Bp + 16);
            v16bf b = __builtin_bit_cast(v16bf, ub.v);
            acc[sub] = __builtin_amdgcn_wmma_f32_16x16x32_bf16(
                /*neg_a=*/false, a, /*neg_b=*/false, b,
                /*c_mod=*/(short)0, acc[sub], /*reuse_a=*/false, /*reuse_b=*/false);
        }
    }

    #pragma unroll
    for (int sub = 0; sub < 4; ++sub) {
        const int ncol = n0 + sub * 16 + ln;
        const float bb = bias[ncol];
        #pragma unroll
        for (int v = 0; v < 8; ++v) {
            int m = m0 + hs * 8 + v;   // D layout: lanes 0-15 -> M 0-7, 16-31 -> M 8-15
            if (m < Nrows) C[(size_t)m * Ncols + ncol] = acc[sub][v] + bb;
        }
    }
}

// ---- per-layer scratch init ----------------------------------------------

__global__ void layer_init(float* __restrict__ agg, unsigned* __restrict__ mEnc,
                           float* __restrict__ denom, int N) {
    int i = blockIdx.x * blockDim.x + threadIdx.x;
    if (i < N * HIDDEN) agg[i] = 0.0f;
    if (i < N * HEADS) { mEnc[i] = NEG_INF_ENC; denom[i] = 0.0f; }
}

// ---- edge attention -------------------------------------------------------

// scores[e,h] = <q[tgt,h,:], k[src,h,:]> / 4 ; running segment max via atomicMax
__global__ void edge_scores(const float* __restrict__ qkvr,
                            const long long* __restrict__ src, const long long* __restrict__ tgt,
                            float* __restrict__ scoreP, unsigned* __restrict__ mEnc, int E) {
    int idx = blockIdx.x * blockDim.x + threadIdx.x;
    if (idx >= E * HEADS) return;
    int e = idx >> 3, hh = idx & 7;
    long long s = src[e], t = tgt[e];
    const float* q = qkvr + (size_t)t * 512 + hh * HEAD_DIM;        // Q at offset 0
    const float* k = qkvr + (size_t)s * 512 + 128 + hh * HEAD_DIM;  // K at offset 128
    float acc = 0.0f;
    #pragma unroll
    for (int d = 0; d < HEAD_DIM; d += 4) {
        float4 qa = *(const float4*)(q + d), ka = *(const float4*)(k + d);
        acc += qa.x * ka.x + qa.y * ka.y + qa.z * ka.z + qa.w * ka.w;
    }
    acc *= 0.25f;                                                   // 1/sqrt(16)
    scoreP[idx] = acc;
    atomicMax(&mEnc[(size_t)t * HEADS + hh], encF(acc));
}

__global__ void finalize_max(unsigned* __restrict__ mEnc, int N) {
    int i = blockIdx.x * blockDim.x + threadIdx.x;
    if (i >= N * HEADS) return;
    float m = decF(mEnc[i]);
    if (!isfinite(m)) m = 0.0f;
    ((float*)mEnc)[i] = m;      // buffer becomes float m from here on
}

__global__ void edge_exp(const float* __restrict__ mF, const long long* __restrict__ tgt,
                         float* __restrict__ scoreP, float* __restrict__ denom, int E) {
    int idx = blockIdx.x * blockDim.x + threadIdx.x;
    if (idx >= E * HEADS) return;
    int e = idx >> 3, hh = idx & 7;
    long long t = tgt[e];
    float p = expf(scoreP[idx] - mF[(size_t)t * HEADS + hh]);
    scoreP[idx] = p;
    atomicAdd(&denom[(size_t)t * HEADS + hh], p);
}

__global__ void edge_agg(const float* __restrict__ qkvr,
                         const long long* __restrict__ src, const long long* __restrict__ tgt,
                         const float* __restrict__ scoreP, const float* __restrict__ denom,
                         float* __restrict__ agg, int E) {
    int idx = blockIdx.x * blockDim.x + threadIdx.x;
    if (idx >= E * HEADS) return;
    int e = idx >> 3, hh = idx & 7;
    long long s = src[e], t = tgt[e];
    float w = scoreP[idx] / fmaxf(denom[(size_t)t * HEADS + hh], 1e-16f);
    const float* v = qkvr + (size_t)s * 512 + 256 + hh * HEAD_DIM;  // V at offset 256
    float* a = agg + (size_t)t * HIDDEN + hh * HEAD_DIM;
    #pragma unroll
    for (int d = 0; d < HEAD_DIM; ++d) atomicAdd(a + d, w * v[d]);
}

// ---- beta gate + residual + LayerNorm (one 128-thread block per node) -----

__global__ __launch_bounds__(128)
void node_update(const float* __restrict__ agg, const float* __restrict__ qkvr,
                 const float* __restrict__ Wbeta, const float* __restrict__ ln_g,
                 const float* __restrict__ ln_b, float* __restrict__ h, int N) {
    int node = blockIdx.x;
    if (node >= N) return;
    int tid = threadIdx.x;
    __shared__ float red[4];

    float outv = agg[(size_t)node * HIDDEN + tid];
    float rv   = qkvr[(size_t)node * 512 + 384 + tid];              // skip at offset 384

    float part = outv * Wbeta[tid] + rv * Wbeta[HIDDEN + tid]
               + (outv - rv) * Wbeta[2 * HIDDEN + tid];
    float logit = blockSum128(part, red);
    float beta  = 1.0f / (1.0f + expf(-logit));
    float hn    = beta * rv + (1.0f - beta) * outv;

    float y   = h[(size_t)node * HIDDEN + tid] + hn;
    float mu  = blockSum128(y, red) * (1.0f / HIDDEN);
    float d   = y - mu;
    float var = blockSum128(d * d, red) * (1.0f / HIDDEN);
    h[(size_t)node * HIDDEN + tid] = d * rsqrtf(var + 1e-5f) * ln_g[tid] + ln_b[tid];
}

// ---- graph pooling --------------------------------------------------------

__global__ void pool_init(float* __restrict__ gsum, unsigned* __restrict__ gmax,
                          int* __restrict__ gcnt) {
    int i = blockIdx.x * blockDim.x + threadIdx.x;
    if (i < NGRAPHS * HIDDEN) { gsum[i] = 0.0f; gmax[i] = NEG_INF_ENC; }
    if (i < NGRAPHS) gcnt[i] = 0;
}

__global__ void pool_accum(const float* __restrict__ h, const long long* __restrict__ batch,
                           float* __restrict__ gsum, unsigned* __restrict__ gmax,
                           int* __restrict__ gcnt, int N) {
    int i = blockIdx.x * blockDim.x + threadIdx.x;
    if (i >= N * HIDDEN) return;
    int node = i >> 7, d = i & 127;
    int g = (int)batch[node];
    float v = h[i];
    atomicAdd(&gsum[g * HIDDEN + d], v);
    atomicMax(&gmax[g * HIDDEN + d], encF(v));
    if (d == 0) atomicAdd(&gcnt[g], 1);
}

__global__ void pool_finalize(const float* __restrict__ gsum, const unsigned* __restrict__ gmax,
                              const int* __restrict__ gcnt, float* __restrict__ out) {
    int i = blockIdx.x * blockDim.x + threadIdx.x;
    if (i >= NGRAPHS * HIDDEN) return;
    int g = i >> 7;
    float cnt = fmaxf((float)gcnt[g], 1.0f);
    out[i] = gsum[i] / cnt;                              // local_repr
    float m = decF(gmax[i]);
    out[NGRAPHS * HIDDEN + i] = isfinite(m) ? m : 0.0f;  // global_repr
}

__global__ void copy_h(const float* __restrict__ h, float* __restrict__ out, int N) {
    int i = blockIdx.x * blockDim.x + threadIdx.x;
    if (i < N * HIDDEN) out[2 * NGRAPHS * HIDDEN + i] = h[i];
}

// ---------------------------------------------------------------------------

extern "C" void kernel_launch(void* const* d_in, const int* in_sizes, int n_in,
                              void* d_out, int out_size, void* d_ws, size_t ws_size,
                              hipStream_t stream) {
    const float*     x     = (const float*)d_in[0];
    const long long* ei    = (const long long*)d_in[1];
    const long long* batch = (const long long*)d_in[2];
    const float*     Win   = (const float*)d_in[3];
    const float*     b_in  = (const float*)d_in[4];
    const float*     Wq    = (const float*)d_in[5];
    const float*     bq    = (const float*)d_in[6];
    const float*     Wk    = (const float*)d_in[7];
    const float*     bk    = (const float*)d_in[8];
    const float*     Wv    = (const float*)d_in[9];
    const float*     bv    = (const float*)d_in[10];
    const float*     Wsk   = (const float*)d_in[11];
    const float*     bsk   = (const float*)d_in[12];
    const float*     Wbeta = (const float*)d_in[13];
    const float*     ln_g  = (const float*)d_in[14];
    const float*     ln_b  = (const float*)d_in[15];

    const int N    = in_sizes[0] / 64;        // nodes (F_IN = 64)
    const int E    = in_sizes[1] / 2;         // edges
    const int Mpad = ((N + 15) / 16) * 16;    // padded rows for guard-free WMMA loads
    const long long* src = ei;
    const long long* tgt = ei + E;

    char* wsp = (char*)d_ws;
    auto carve = [&](size_t bytes) {
        char* p = wsp;
        wsp += (bytes + 255) & ~(size_t)255;
        return p;
    };
    float*          h      = (float*)carve((size_t)N * HIDDEN * 4);
    float*          qkvr   = (float*)carve((size_t)N * 512 * 4);
    float*          agg    = (float*)carve((size_t)N * HIDDEN * 4);
    unsigned*       mEnc   = (unsigned*)carve((size_t)N * HEADS * 4);
    float*          denom  = (float*)carve((size_t)N * HEADS * 4);
    float*          scoreP = (float*)carve((size_t)E * HEADS * 4);
    unsigned short* xb     = (unsigned short*)carve((size_t)Mpad * 64 * 2);
    unsigned short* hb     = (unsigned short*)carve((size_t)Mpad * HIDDEN * 2);
    unsigned short* WinB   = (unsigned short*)carve((size_t)HIDDEN * 64 * 2);
    unsigned short* WallB  = (unsigned short*)carve((size_t)NLAYERS * 512 * HIDDEN * 2);
    float*          ball   = (float*)carve((size_t)NLAYERS * 512 * 4);
    float*          gsum   = (float*)carve((size_t)NGRAPHS * HIDDEN * 4);
    unsigned*       gmax   = (unsigned*)carve((size_t)NGRAPHS * HIDDEN * 4);
    int*            gcnt   = (int*)carve((size_t)NGRAPHS * 4);

    const int TB = 256;
    auto blocks = [&](long long n) { return (unsigned)((n + TB - 1) / TB); };

    // weight prep (bf16)
    conv_bf16_pad<<<blocks(HIDDEN * 64), TB, 0, stream>>>(Win, WinB,
                                                          HIDDEN * 64, HIDDEN * 64);
    prep_wall<<<blocks((long long)NLAYERS * 512 * HIDDEN), TB, 0, stream>>>(Wq, Wk, Wv, Wsk, WallB);
    prep_bias<<<blocks(NLAYERS * 512), TB, 0, stream>>>(bq, bk, bv, bsk, ball);

    // input projection: h = x @ Win^T + b_in   (K = 64)
    conv_bf16_pad<<<blocks((long long)Mpad * 64), TB, 0, stream>>>(
        x, xb, (long long)N * 64, (long long)Mpad * 64);
    {
        dim3 g(Mpad / 16, HIDDEN / 64);
        gemm_bf16_wmma<64><<<g, 32, 0, stream>>>(xb, WinB, b_in, h, N, HIDDEN);
    }

    for (int l = 0; l < NLAYERS; ++l) {
        // Q|K|V|skip in one fused GEMM: qkvr = h @ Wall^T + ball (K=128, Ncols=512)
        conv_bf16_pad<<<blocks((long long)Mpad * HIDDEN), TB, 0, stream>>>(
            h, hb, (long long)N * HIDDEN, (long long)Mpad * HIDDEN);
        dim3 g(Mpad / 16, 512 / 64);
        gemm_bf16_wmma<128><<<g, 32, 0, stream>>>(
            hb, WallB + (size_t)l * 512 * HIDDEN, ball + l * 512, qkvr, N, 512);

        layer_init  <<<blocks((long long)N * HIDDEN), TB, 0, stream>>>(agg, mEnc, denom, N);
        edge_scores <<<blocks((long long)E * HEADS), TB, 0, stream>>>(qkvr, src, tgt, scoreP, mEnc, E);
        finalize_max<<<blocks((long long)N * HEADS), TB, 0, stream>>>(mEnc, N);
        edge_exp    <<<blocks((long long)E * HEADS), TB, 0, stream>>>((const float*)mEnc, tgt, scoreP, denom, E);
        edge_agg    <<<blocks((long long)E * HEADS), TB, 0, stream>>>(qkvr, src, tgt, scoreP, denom, agg, E);
        node_update <<<N, 128, 0, stream>>>(agg, qkvr, Wbeta + (size_t)l * 3 * HIDDEN,
                                            ln_g + l * HIDDEN, ln_b + l * HIDDEN, h, N);
    }

    float* out = (float*)d_out;
    pool_init    <<<blocks(NGRAPHS * HIDDEN), TB, 0, stream>>>(gsum, gmax, gcnt);
    pool_accum   <<<blocks((long long)N * HIDDEN), TB, 0, stream>>>(h, batch, gsum, gmax, gcnt, N);
    pool_finalize<<<blocks(NGRAPHS * HIDDEN), TB, 0, stream>>>(gsum, gmax, gcnt, out);
    copy_h       <<<blocks((long long)N * HIDDEN), TB, 0, stream>>>(h, out, N);
}
